// EncoderMambaDecoderBlock_28793460752848
// MI455X (gfx1250) — matrix operand order repaired and checked by reading, not compiled, not run on hardware
//
#include <hip/hip_runtime.h>

typedef _Float16 f16;
typedef __attribute__((ext_vector_type(16))) _Float16 v16h;
typedef __attribute__((ext_vector_type(2)))  _Float16 h2v;
typedef __attribute__((ext_vector_type(8)))  float    v8f;

// Problem dimensions (fixed by the reference)
static constexpr int BB   = 8;
static constexpr int LL   = 2048;
static constexpr int NT   = BB * LL;      // 16384 tokens
static constexpr int DIN  = 64;
static constexpr int HH   = 256;
static constexpr int DM   = 512;
static constexpr int DI   = 1024;         // d_inner
static constexpr int DS   = 16;           // d_state
static constexpr int DTR  = 32;           // dt_rank
static constexpr int DOUT = 64;

// ---------------------------------------------------------------------------
// f32 -> f16 conversion (weights + network input, one-time per launch)
// ---------------------------------------------------------------------------
__global__ void cvt_f32_f16(const float* __restrict__ s, f16* __restrict__ d, int n) {
    int i = blockIdx.x * blockDim.x + threadIdx.x;
    if (i < n) d[i] = (f16)s[i];
}

// ---------------------------------------------------------------------------
// WMMA GEMM:  C[M,N] = act( A[M,K](lda) * W[N,K]^T + bias[N] )
//
// Register-blocked: one wave32 owns a 32x64 tile of C = 2 (M) x 4 (N) WMMA
// tiles -> 8 v_wmma_f32_16x16x32_f16 per K-step from 6 fragment loads
// (A fragments reused across 4 N-tiles, B fragments across 2 M-tiles).
//
// Fragment layouts per CDNA5 ISA 7.12.2 (wave32):
//   A (16x32 f16): lane row M = lane&15, half-sel h = lane>>4,
//                  VGPR j<4 -> K = {8h+2j, 8h+2j+1}, j>=4 -> K = {16+8h+2(j-4), ...}
//   B (32x16 f16): lane col N = lane&15, same K mapping as A
//   C/D (16x16 f32): VGPR r -> M = r + 8h, N = lane&15
// ---------------------------------------------------------------------------
enum { EPI_NONE = 0, EPI_BIAS = 1, EPI_RELU = 2, EPI_SOFTPLUS = 3 };

template <int EPI, bool OUTF32>
__global__ __launch_bounds__(256)
void gemm_f16_wmma(const f16* __restrict__ A, int lda,
                   const f16* __restrict__ W,            // (N,K) row-major, ldw == K
                   const float* __restrict__ bias,
                   f16* __restrict__ Ch, float* __restrict__ Cf,
                   int ldc, int M, int N, int K)
{
    int wave  = (int)((blockIdx.x * blockDim.x + threadIdx.x) >> 5);
    int lane  = threadIdx.x & 31;
    int tilesN = N >> 6;                        // 64-wide N strips
    int tm = wave / tilesN;                     // 32-tall M strip
    int tn = wave - tm * tilesN;
    if ((tm << 5) >= M) return;                 // uniform per wave

    int l15  = lane & 15;
    int hsel = lane >> 4;

    const f16* __restrict__ arow0 = A + (size_t)((tm << 5) + l15) * lda;
    const f16* __restrict__ arow1 = arow0 + (size_t)16 * lda;
    const f16* __restrict__ brow0 = W + (size_t)((tn << 6) +  0 + l15) * K;
    const f16* __restrict__ brow1 = W + (size_t)((tn << 6) + 16 + l15) * K;
    const f16* __restrict__ brow2 = W + (size_t)((tn << 6) + 32 + l15) * K;
    const f16* __restrict__ brow3 = W + (size_t)((tn << 6) + 48 + l15) * K;

    v8f acc[2][4];
#pragma unroll
    for (int p = 0; p < 2; ++p)
#pragma unroll
        for (int q = 0; q < 4; ++q) acc[p][q] = (v8f){};

    for (int k0 = 0; k0 < K; k0 += 32) {
        // hide latency of the next K-slab (speculative, OOB-safe)
        __builtin_prefetch(arow0 + k0 + 128, 0, 0);
        __builtin_prefetch(arow1 + k0 + 128, 0, 0);
        v16h a0, a1, b0, b1, b2, b3;
#pragma unroll
        for (int j = 0; j < 8; ++j) {
            int kp = (j < 4) ? (8 * hsel + 2 * j) : (16 + 8 * hsel + 2 * (j - 4));
            h2v p0 = *(const h2v*)(arow0 + k0 + kp);
            h2v p1 = *(const h2v*)(arow1 + k0 + kp);
            a0[2 * j] = p0.x; a0[2 * j + 1] = p0.y;
            a1[2 * j] = p1.x; a1[2 * j + 1] = p1.y;
            int kb = kp;
            h2v q0 = *(const h2v*)(brow0 + k0 + kb);
            h2v q1 = *(const h2v*)(brow1 + k0 + kb);
            h2v q2 = *(const h2v*)(brow2 + k0 + kb);
            h2v q3 = *(const h2v*)(brow3 + k0 + kb);
            b0[2 * j] = q0.x; b0[2 * j + 1] = q0.y;
            b1[2 * j] = q1.x; b1[2 * j + 1] = q1.y;
            b2[2 * j] = q2.x; b2[2 * j + 1] = q2.y;
            b3[2 * j] = q3.x; b3[2 * j + 1] = q3.y;
        }
        acc[0][0] = __builtin_amdgcn_wmma_f32_16x16x32_f16(false, a0, false, b0, (short)0, acc[0][0], false, false);
        acc[1][0] = __builtin_amdgcn_wmma_f32_16x16x32_f16(false, a1, false, b0, (short)0, acc[1][0], false, false);
        acc[0][1] = __builtin_amdgcn_wmma_f32_16x16x32_f16(false, a0, false, b1, (short)0, acc[0][1], false, false);
        acc[1][1] = __builtin_amdgcn_wmma_f32_16x16x32_f16(false, a1, false, b1, (short)0, acc[1][1], false, false);
        acc[0][2] = __builtin_amdgcn_wmma_f32_16x16x32_f16(false, a0, false, b2, (short)0, acc[0][2], false, false);
        acc[1][2] = __builtin_amdgcn_wmma_f32_16x16x32_f16(false, a1, false, b2, (short)0, acc[1][2], false, false);
        acc[0][3] = __builtin_amdgcn_wmma_f32_16x16x32_f16(false, a0, false, b3, (short)0, acc[0][3], false, false);
        acc[1][3] = __builtin_amdgcn_wmma_f32_16x16x32_f16(false, a1, false, b3, (short)0, acc[1][3], false, false);
    }

#pragma unroll
    for (int q = 0; q < 4; ++q) {
        int col = (tn << 6) + (q << 4) + l15;
        float bv = 0.f;
        if (EPI != EPI_NONE) bv = bias[col];
#pragma unroll
        for (int p = 0; p < 2; ++p) {
#pragma unroll
            for (int r = 0; r < 8; ++r) {
                int row = (tm << 5) + (p << 4) + (hsel << 3) + r;
                float v = acc[p][q][r] + bv;
                if (EPI == EPI_RELU)     v = v > 0.f ? v : 0.f;
                if (EPI == EPI_SOFTPLUS) v = (v > 20.f) ? v : logf(1.f + __expf(v));
                if (OUTF32) Cf[(size_t)row * ldc + col] = v;
                else        Ch[(size_t)row * ldc + col] = (f16)v;
            }
        }
    }
}

// ---------------------------------------------------------------------------
// Depthwise causal conv1d (4 taps, left pad 3) + bias + SiLU.
// One thread per (batch, channel); coalesced across channels each step.
// Reads xm half of xz (stride 2*DI), writes xm_f16 (stride DI).
// ---------------------------------------------------------------------------
__global__ __launch_bounds__(256)
void conv_silu_kernel(const f16* __restrict__ xz, const float* __restrict__ cw,
                      const float* __restrict__ cb, f16* __restrict__ xm)
{
    int tid = blockIdx.x * blockDim.x + threadIdx.x;
    if (tid >= BB * DI) return;
    int b = tid / DI, d = tid - b * DI;

    float w0 = cw[d * 4 + 0], w1 = cw[d * 4 + 1];
    float w2 = cw[d * 4 + 2], w3 = cw[d * 4 + 3];
    float bias = cb[d];
    float x0 = 0.f, x1 = 0.f, x2 = 0.f;

    const f16* __restrict__ src = xz + (size_t)b * LL * (2 * DI) + d;
    f16* __restrict__ dst = xm + (size_t)b * LL * DI + d;
    for (int t = 0; t < LL; ++t) {
        float xt = (float)src[(size_t)t * (2 * DI)];
        float v  = w0 * x0 + w1 * x1 + w2 * x2 + w3 * xt + bias;
        float s  = v / (1.f + __expf(-v));
        dst[(size_t)t * DI] = (f16)s;
        x0 = x1; x1 = x2; x2 = xt;
    }
}

// ---------------------------------------------------------------------------
// Selective scan, fused with D skip-connection and SiLU(z) gating.
// Block = 256 channels of one batch; grid = BB * (DI/256) = 32 blocks.
// State h[16] lives in registers; B_t/C_t (shared by all channels of a batch)
// staged through LDS in 64-token chunks.
// ---------------------------------------------------------------------------
__global__ __launch_bounds__(256)
void scan_kernel(const f16* __restrict__ xm, const f16* __restrict__ delta,
                 const f16* __restrict__ dbl, const f16* __restrict__ xz,
                 const float* __restrict__ A_log, const float* __restrict__ Dp,
                 f16* __restrict__ y)
{
    __shared__ f16 sh[64 * 32];                  // 64 tokens x (16 B + 16 C)
    int b = blockIdx.x >> 2;
    int d = ((blockIdx.x & 3) << 8) + threadIdx.x;

    float Arow[DS];
#pragma unroll
    for (int s = 0; s < DS; ++s) Arow[s] = -__expf(A_log[d * DS + s]);
    float Dd = Dp[d];
    float hst[DS];
#pragma unroll
    for (int s = 0; s < DS; ++s) hst[s] = 0.f;

    size_t base = (size_t)b * LL;
    for (int t0 = 0; t0 < LL; t0 += 64) {
        __syncthreads();
        for (int i = threadIdx.x; i < 64 * 32; i += 256) {
            int tl = i >> 5, c = i & 31;
            sh[i] = dbl[(base + t0 + tl) * 64 + 32 + c];
        }
        __syncthreads();
        for (int tl = 0; tl < 64; ++tl) {
            size_t row = base + t0 + tl;
            float x  = (float)xm[row * DI + d];
            float dl = (float)delta[row * DI + d];
            float dx = dl * x;
            float yv = 0.f;
#pragma unroll
            for (int s = 0; s < DS; ++s) {
                float dA = __expf(dl * Arow[s]);
                hst[s] = dA * hst[s] + dx * (float)sh[tl * 32 + s];
                yv += hst[s] * (float)sh[tl * 32 + 16 + s];
            }
            float z = (float)xz[row * (2 * DI) + DI + d];
            float g = z / (1.f + __expf(-z));
            y[row * DI + d] = (f16)((yv + x * Dd) * g);
        }
    }
}

// ---------------------------------------------------------------------------
// Host side
// ---------------------------------------------------------------------------
static inline char* ws_take(void* ws, size_t& off, size_t bytes) {
    char* p = (char*)ws + off;
    off = (off + bytes + 255) & ~(size_t)255;
    return p;
}

extern "C" void kernel_launch(void* const* d_in, const int* in_sizes, int n_in,
                              void* d_out, int out_size, void* d_ws, size_t ws_size,
                              hipStream_t stream)
{
    (void)in_sizes; (void)n_in; (void)out_size; (void)ws_size;
    const float* x_f32      = (const float*)d_in[0];
    const float* enc_w1     = (const float*)d_in[1];
    const float* enc_b1     = (const float*)d_in[2];
    const float* enc_w2     = (const float*)d_in[3];
    const float* enc_b2     = (const float*)d_in[4];
    const float* in_proj_w  = (const float*)d_in[5];
    const float* conv_w     = (const float*)d_in[6];
    const float* conv_b     = (const float*)d_in[7];
    const float* x_proj_w   = (const float*)d_in[8];
    const float* dt_proj_w  = (const float*)d_in[9];
    const float* dt_proj_b  = (const float*)d_in[10];
    const float* A_log      = (const float*)d_in[11];
    const float* D_param    = (const float*)d_in[12];
    const float* out_proj_w = (const float*)d_in[13];
    const float* dec_w1     = (const float*)d_in[14];
    const float* dec_b1     = (const float*)d_in[15];
    const float* dec_w2     = (const float*)d_in[16];
    const float* dec_b2     = (const float*)d_in[17];
    float* out_f32 = (float*)d_out;

    size_t off = 0;
    f16* xh        = (f16*)ws_take(d_ws, off, sizeof(f16) * (size_t)NT * DIN);
    f16* wh_enc1   = (f16*)ws_take(d_ws, off, sizeof(f16) * (size_t)HH * DIN);
    f16* wh_enc2   = (f16*)ws_take(d_ws, off, sizeof(f16) * (size_t)DM * HH);
    f16* wh_inproj = (f16*)ws_take(d_ws, off, sizeof(f16) * (size_t)2 * DI * DM);
    f16* wh_xproj  = (f16*)ws_take(d_ws, off, sizeof(f16) * (size_t)(DTR + 2 * DS) * DI);
    f16* wh_dtproj = (f16*)ws_take(d_ws, off, sizeof(f16) * (size_t)DI * DTR);
    f16* wh_outprj = (f16*)ws_take(d_ws, off, sizeof(f16) * (size_t)DM * DI);
    f16* wh_dec1   = (f16*)ws_take(d_ws, off, sizeof(f16) * (size_t)HH * DM);
    f16* wh_dec2   = (f16*)ws_take(d_ws, off, sizeof(f16) * (size_t)DOUT * HH);
    f16* h1        = (f16*)ws_take(d_ws, off, sizeof(f16) * (size_t)NT * HH);
    f16* u         = (f16*)ws_take(d_ws, off, sizeof(f16) * (size_t)NT * DM);
    f16* xz        = (f16*)ws_take(d_ws, off, sizeof(f16) * (size_t)NT * 2 * DI);
    f16* xm        = (f16*)ws_take(d_ws, off, sizeof(f16) * (size_t)NT * DI);
    f16* dbl       = (f16*)ws_take(d_ws, off, sizeof(f16) * (size_t)NT * (DTR + 2 * DS));
    f16* delta     = (f16*)ws_take(d_ws, off, sizeof(f16) * (size_t)NT * DI);
    f16* yb        = (f16*)ws_take(d_ws, off, sizeof(f16) * (size_t)NT * DI);
    f16* outb      = (f16*)ws_take(d_ws, off, sizeof(f16) * (size_t)NT * DM);
    f16* h2b       = (f16*)ws_take(d_ws, off, sizeof(f16) * (size_t)NT * HH);

    auto cvt = [&](const float* s, f16* d, int n) {
        cvt_f32_f16<<<(n + 255) / 256, 256, 0, stream>>>(s, d, n);
    };
    cvt(x_f32,      xh,        NT * DIN);
    cvt(enc_w1,     wh_enc1,   HH * DIN);
    cvt(enc_w2,     wh_enc2,   DM * HH);
    cvt(in_proj_w,  wh_inproj, 2 * DI * DM);
    cvt(x_proj_w,   wh_xproj,  (DTR + 2 * DS) * DI);
    cvt(dt_proj_w,  wh_dtproj, DI * DTR);
    cvt(out_proj_w, wh_outprj, DM * DI);
    cvt(dec_w1,     wh_dec1,   HH * DM);
    cvt(dec_w2,     wh_dec2,   DOUT * HH);

    // one wave per 32x64 C-tile, 8 waves per block
    auto gblk = [](int M, int N) { return ((M / 32) * (N / 64)) / 8; };

    // enc1: h1 = relu(x @ enc_w1^T + b1)           [16384 x 256 x 64]
    gemm_f16_wmma<EPI_RELU, false><<<gblk(NT, HH), 256, 0, stream>>>(
        xh, DIN, wh_enc1, enc_b1, h1, nullptr, HH, NT, HH, DIN);
    // enc2: u = h1 @ enc_w2^T + b2                 [16384 x 512 x 256]
    gemm_f16_wmma<EPI_BIAS, false><<<gblk(NT, DM), 256, 0, stream>>>(
        h1, HH, wh_enc2, enc_b2, u, nullptr, DM, NT, DM, HH);
    // in_proj: xz = u @ in_proj^T                  [16384 x 2048 x 512]
    gemm_f16_wmma<EPI_NONE, false><<<gblk(NT, 2 * DI), 256, 0, stream>>>(
        u, DM, wh_inproj, nullptr, xz, nullptr, 2 * DI, NT, 2 * DI, DM);
    // depthwise causal conv + bias + SiLU -> xm
    conv_silu_kernel<<<(BB * DI) / 256, 256, 0, stream>>>(xz, conv_w, conv_b, xm);
    // x_proj: dbl = xm @ x_proj^T                  [16384 x 64 x 1024]
    gemm_f16_wmma<EPI_NONE, false><<<gblk(NT, DTR + 2 * DS), 256, 0, stream>>>(
        xm, DI, wh_xproj, nullptr, dbl, nullptr, DTR + 2 * DS, NT, DTR + 2 * DS, DI);
    // dt_proj: delta = softplus(dt @ dt_proj^T + b)  (A = dbl with lda=64, K=32)
    gemm_f16_wmma<EPI_SOFTPLUS, false><<<gblk(NT, DI), 256, 0, stream>>>(
        dbl, DTR + 2 * DS, wh_dtproj, dt_proj_b, delta, nullptr, DI, NT, DI, DTR);
    // selective scan + D skip + SiLU(z) gate -> y
    scan_kernel<<<BB * (DI / 256), 256, 0, stream>>>(
        xm, delta, dbl, xz, A_log, D_param, yb);
    // out_proj: out = y @ out_proj^T               [16384 x 512 x 1024]
    gemm_f16_wmma<EPI_NONE, false><<<gblk(NT, DM), 256, 0, stream>>>(
        yb, DI, wh_outprj, nullptr, outb, nullptr, DM, NT, DM, DI);
    // dec1: h2 = relu(out @ dec_w1^T + b1)         [16384 x 256 x 512]
    gemm_f16_wmma<EPI_RELU, false><<<gblk(NT, HH), 256, 0, stream>>>(
        outb, DM, wh_dec1, dec_b1, h2b, nullptr, HH, NT, HH, DM);
    // dec2: d_out(f32) = h2 @ dec_w2^T + b2        [16384 x 64 x 256]
    gemm_f16_wmma<EPI_BIAS, true><<<gblk(NT, DOUT), 256, 0, stream>>>(
        h2b, HH, wh_dec2, dec_b2, nullptr, out_f32, DOUT, NT, DOUT, HH);
}
